// MultiheadAttention_41549513622192
// MI455X (gfx1250) — matrix operand order repaired and checked by reading, not compiled, not run on hardware
//
#include <hip/hip_runtime.h>
#include <hip/hip_bf16.h>

#define B_  4
#define S_  2048
#define D_  1024
#define H_  16
#define HD_ 64

typedef __attribute__((ext_vector_type(16))) _Float16 v16h;
typedef __attribute__((ext_vector_type(8)))  _Float16 v8h;
typedef __attribute__((ext_vector_type(8)))  float    v8f;

// ---------------- WMMA helpers (CDNA5 wave32, 16x16x32 f16 -> f32) ----------------

__device__ __forceinline__ v8f wmma16(v16h a, v16h b, v8f c) {
  // (neg_a, A, neg_b, B, c_mod, C, reuse_a, reuse_b)
  return __builtin_amdgcn_wmma_f32_16x16x32_f16(false, a, false, b, (short)0, c, false, false);
}

// A-matrix 16x32 fragment: lane m = L&15; kbase = (L&16)?16:0;
// elements 0..15 = K kbase..kbase+15 (matches B fragment order).
__device__ __forceinline__ v16h load_a16x32(const _Float16* tile, int ld, int lane) {
  const int m  = lane & 15;
  const int kb = (lane & 16) ? 16 : 0;
  const _Float16* p = tile + m * ld + kb;
  union { v16h v; v8h h[2]; } u;
  u.h[0] = *(const v8h*)(p);
  u.h[1] = *(const v8h*)(p + 8);
  return u.v;
}

// B-matrix 32x16 fragment: memory laid out [n][k] (k contiguous, stride ld between
// columns). lane n = L&15; khalf = (L&16)?16:0; elements 0..15 = K khalf..khalf+15.
__device__ __forceinline__ v16h load_b32x16(const _Float16* colmaj, int ld, int lane) {
  const int n  = lane & 15;
  const int kh = (lane & 16) ? 16 : 0;
  return *(const v16h*)(colmaj + n * ld + kh);
}

// ---------------- fp32 -> f16 convert ----------------

__global__ void f32_to_f16_kernel(const float* __restrict__ src,
                                  _Float16* __restrict__ dst, int n) {
  int i = blockIdx.x * blockDim.x + threadIdx.x;
  if (i < n) dst[i] = (_Float16)src[i];
}

// ---------------- QKV projection GEMM ----------------
// C[8192,3072] = A[8192,1024] @ W^T, + bias. Block tile 128x128; 8 waves as 4x2,
// each wave 32x64 (2x4 WMMA frags, 8 WMMAs per K-step). Epilogue scatters into
// q16/k16 [B,H,S,HD] and vT16 [B,H,HD,S]; Q gets the 1/sqrt(HD) scale folded in.
__global__ __launch_bounds__(256)
void qkv_gemm_kernel(const _Float16* __restrict__ A,
                     const _Float16* __restrict__ W,     // [3072,1024] row-major
                     const float*    __restrict__ bias,  // [3072]
                     _Float16* __restrict__ q16,
                     _Float16* __restrict__ k16,
                     _Float16* __restrict__ vT16) {
  const int lane    = threadIdx.x & 31;
  const int wave    = threadIdx.x >> 5;            // 0..7
  const int rowBase = blockIdx.y * 128 + (wave >> 1) * 32;
  const int colBase = blockIdx.x * 128 + (wave & 1) * 64;

  const v8f zero = {0.f,0.f,0.f,0.f,0.f,0.f,0.f,0.f};
  v8f acc[2][4];
#pragma unroll
  for (int i = 0; i < 2; ++i)
#pragma unroll
    for (int j = 0; j < 4; ++j) acc[i][j] = zero;

#pragma unroll 2
  for (int k0 = 0; k0 < D_; k0 += 32) {
    v16h af[2], bf[4];
#pragma unroll
    for (int i = 0; i < 2; ++i)
      af[i] = load_a16x32(A + (size_t)(rowBase + 16 * i) * D_ + k0, D_, lane);
#pragma unroll
    for (int j = 0; j < 4; ++j)
      bf[j] = load_b32x16(W + (size_t)(colBase + 16 * j) * D_ + k0, D_, lane);
#pragma unroll
    for (int i = 0; i < 2; ++i)
#pragma unroll
      for (int j = 0; j < 4; ++j)
        acc[i][j] = wmma16(af[i], bf[j], acc[i][j]);
  }

  const int half8 = (lane & 16) ? 8 : 0;
  const int n     = lane & 15;
#pragma unroll
  for (int j = 0; j < 4; ++j) {
    // cj is wave-uniform (a 16-wide span never crosses a 1024 boundary);
    // force it scalar so the Q/K/V select becomes one uniform branch per j.
    const int   cj     = __builtin_amdgcn_readfirstlane(colBase + 16 * j);
    const int   which  = cj >> 10;            // 0=Q 1=K 2=V (scalar)
    const int   h      = (cj & 1023) >> 6;    // scalar
    const int   hdBase = cj & 63;             // scalar, + n < 64
    const float bj     = bias[cj + n];
#pragma unroll
    for (int i = 0; i < 2; ++i) {
#pragma unroll
      for (int v = 0; v < 8; ++v) {
        const int   mrow = rowBase + 16 * i + v + half8;
        const int   b    = mrow >> 11;        // /2048
        const int   s    = mrow & 2047;
        const float val  = acc[i][j][v] + bj;
        const int   hd   = hdBase + n;
        if (which == 0) {
          q16[(((size_t)(b * H_ + h) * S_ + s) * HD_) + hd] = (_Float16)(val * 0.125f);
        } else if (which == 1) {
          k16[(((size_t)(b * H_ + h) * S_ + s) * HD_) + hd] = (_Float16)val;
        } else {
          vT16[(((size_t)(b * H_ + h) * HD_ + hd) * S_) + s] = (_Float16)val;
        }
      }
    }
  }
}

// ---------------- Flash attention ----------------
// grid (B*H, S/64); 128 threads = 4 waves, each wave owns 16 query rows.
// KV tiles of 32 staged in LDS with GLOBAL_LOAD_ASYNC_TO_LDS (ASYNCcnt-tracked),
// shared by all 4 waves.
__global__ __launch_bounds__(128)
void attn_kernel(const _Float16* __restrict__ q16,
                 const _Float16* __restrict__ k16,
                 const _Float16* __restrict__ vT16,
                 _Float16* __restrict__ ctx16) {   // [B,S,D]
  __shared__ alignas(64) _Float16 Kt[32][64];      // [kv][hd] (4 KB, contiguous in k16)
  __shared__ alignas(64) _Float16 Vt[64][32];      // [hd][kv]
  __shared__ alignas(64) _Float16 Pb[4][16][32];   // per-wave P staging

  const int lane  = threadIdx.x & 31;
  const int wave  = threadIdx.x >> 5;         // 0..3
  const int bh    = blockIdx.x;               // 0..63
  const int b     = bh >> 4;
  const int h     = bh & 15;
  const int q0    = blockIdx.y * 64 + wave * 16;
  const int n     = lane & 15;
  const int half8 = (lane & 16) ? 8 : 0;

  // LDS byte offsets (generic LDS pointer low 32 bits = LDS address)
  const unsigned ktBase = (unsigned)(uintptr_t)(&Kt[0][0]);
  const unsigned vtBase = (unsigned)(uintptr_t)(&Vt[0][0]);

  // Q fragment (already scaled by 1/sqrt(HD)), K slices 0..31 and 32..63
  const _Float16* qbase = q16 + ((size_t)bh * S_ + q0) * HD_;
  const v16h qf0 = load_a16x32(qbase +  0, HD_, lane);
  const v16h qf1 = load_a16x32(qbase + 32, HD_, lane);

  const v8f zero = {0.f,0.f,0.f,0.f,0.f,0.f,0.f,0.f};
  v8f O[4] = {zero, zero, zero, zero};
  float mst[8], lst[8];
#pragma unroll
  for (int v = 0; v < 8; ++v) { mst[v] = -3.0e38f; lst[v] = 0.0f; }

  const int t = threadIdx.x;
  for (int j0 = 0; j0 < S_; j0 += 32) {
    __syncthreads();   // previous iteration done with Kt/Vt
    {
      // K tile: 4 KB fully contiguous run of k16 (rows j0..j0+31, 64 f16 each).
      // 128 threads x 16 B x 2 issues.
      const _Float16* ksrc = k16 + ((size_t)bh * S_ + j0) * HD_;
      unsigned long long ga0 = (unsigned long long)(uintptr_t)(ksrc + t * 8);
      unsigned long long ga1 = (unsigned long long)(uintptr_t)(ksrc + 1024 + t * 8);
      unsigned l0 = ktBase + (unsigned)(t * 16);
      unsigned l1 = ktBase + 2048u + (unsigned)(t * 16);
      asm volatile("global_load_async_to_lds_b128 %0, %1, off"
                   :: "v"(l0), "v"(ga0) : "memory");
      asm volatile("global_load_async_to_lds_b128 %0, %1, off"
                   :: "v"(l1), "v"(ga1) : "memory");
      // V tile: 64 rows (hd) x 32 f16 from vT16; each row a contiguous 64 B run.
      const _Float16* vsrc = vT16 + (size_t)bh * HD_ * S_ + j0;
      const int r0 = t >> 2;            // 0..31
      const int c0 = (t & 3) * 8;       // 0,8,16,24
      unsigned long long gv0 = (unsigned long long)(uintptr_t)(vsrc + (size_t)r0 * S_ + c0);
      unsigned long long gv1 = (unsigned long long)(uintptr_t)(vsrc + (size_t)(r0 + 32) * S_ + c0);
      unsigned lv0 = vtBase + (unsigned)((r0 * 32 + c0) * 2);
      unsigned lv1 = vtBase + (unsigned)(((r0 + 32) * 32 + c0) * 2);
      asm volatile("global_load_async_to_lds_b128 %0, %1, off"
                   :: "v"(lv0), "v"(gv0) : "memory");
      asm volatile("global_load_async_to_lds_b128 %0, %1, off"
                   :: "v"(lv1), "v"(gv1) : "memory");
      asm volatile("s_wait_asynccnt 0x0" ::: "memory");
    }
    __syncthreads();

    // scores: S0 = kv cols 0..15, S1 = kv cols 16..31 (each over hd K=64)
    v8f S0 = zero, S1 = zero;
    S0 = wmma16(qf0, load_b32x16(&Kt[0][0],   64, lane), S0);
    S0 = wmma16(qf1, load_b32x16(&Kt[0][32],  64, lane), S0);
    S1 = wmma16(qf0, load_b32x16(&Kt[16][0],  64, lane), S1);
    S1 = wmma16(qf1, load_b32x16(&Kt[16][32], 64, lane), S1);

    // online softmax, per row (row = v + half8; 16 cols spread over 16 lanes
    // within the same half-wave, so xor masks 1,2,4,8 reduce each row)
    float alpha[8];
#pragma unroll
    for (int v = 0; v < 8; ++v) {
      float mx = fmaxf(S0[v], S1[v]);
#pragma unroll
      for (int off = 8; off > 0; off >>= 1) mx = fmaxf(mx, __shfl_xor(mx, off));
      const float mnew = fmaxf(mst[v], mx);
      const float a    = __expf(mst[v] - mnew);
      const float p0   = __expf(S0[v] - mnew);
      const float p1   = __expf(S1[v] - mnew);
      float rs = p0 + p1;
#pragma unroll
      for (int off = 8; off > 0; off >>= 1) rs += __shfl_xor(rs, off);
      lst[v]   = lst[v] * a + rs;
      mst[v]   = mnew;
      alpha[v] = a;
      S0[v] = p0;
      S1[v] = p1;
    }
#pragma unroll
    for (int nf = 0; nf < 4; ++nf)
#pragma unroll
      for (int v = 0; v < 8; ++v) O[nf][v] *= alpha[v];

    // P (C layout) -> LDS -> A-fragment layout
#pragma unroll
    for (int v = 0; v < 8; ++v) {
      Pb[wave][v + half8][n]      = (_Float16)S0[v];
      Pb[wave][v + half8][16 + n] = (_Float16)S1[v];
    }
    asm volatile("s_wait_dscnt 0" ::: "memory");
    const v16h pf = load_a16x32(&Pb[wave][0][0], 32, lane);

    // O += P @ V  (K = 32 kv positions, N = 64 hd in 4 chunks)
#pragma unroll
    for (int nf = 0; nf < 4; ++nf)
      O[nf] = wmma16(pf, load_b32x16(&Vt[nf * 16][0], 32, lane), O[nf]);
  }

  // normalize and store ctx as [B,S,D] f16
  float inv[8];
#pragma unroll
  for (int v = 0; v < 8; ++v) inv[v] = 1.0f / lst[v];
#pragma unroll
  for (int nf = 0; nf < 4; ++nf) {
#pragma unroll
    for (int v = 0; v < 8; ++v) {
      const int s = q0 + v + half8;
      const int d = h * HD_ + 16 * nf + n;
      ctx16[((size_t)(b * S_) + s) * D_ + d] = (_Float16)(O[nf][v] * inv[v]);
    }
  }
}

// ---------------- Output projection GEMM ----------------
// Same 128x128 block / 32x64 wave tiling as the QKV GEMM; fp32 output + bias.
__global__ __launch_bounds__(256)
void out_gemm_kernel(const _Float16* __restrict__ A,    // ctx16 [8192,1024]
                     const _Float16* __restrict__ W,    // wo16 [1024,1024]
                     const float*    __restrict__ bias, // [1024]
                     float* __restrict__ out) {         // [8192,1024]
  const int lane    = threadIdx.x & 31;
  const int wave    = threadIdx.x >> 5;
  const int rowBase = blockIdx.y * 128 + (wave >> 1) * 32;
  const int colBase = blockIdx.x * 128 + (wave & 1) * 64;

  const v8f zero = {0.f,0.f,0.f,0.f,0.f,0.f,0.f,0.f};
  v8f acc[2][4];
#pragma unroll
  for (int i = 0; i < 2; ++i)
#pragma unroll
    for (int j = 0; j < 4; ++j) acc[i][j] = zero;

#pragma unroll 2
  for (int k0 = 0; k0 < D_; k0 += 32) {
    v16h af[2], bf[4];
#pragma unroll
    for (int i = 0; i < 2; ++i)
      af[i] = load_a16x32(A + (size_t)(rowBase + 16 * i) * D_ + k0, D_, lane);
#pragma unroll
    for (int j = 0; j < 4; ++j)
      bf[j] = load_b32x16(W + (size_t)(colBase + 16 * j) * D_ + k0, D_, lane);
#pragma unroll
    for (int i = 0; i < 2; ++i)
#pragma unroll
      for (int j = 0; j < 4; ++j)
        acc[i][j] = wmma16(af[i], bf[j], acc[i][j]);
  }

  const int half8 = (lane & 16) ? 8 : 0;
  const int n     = lane & 15;
#pragma unroll
  for (int j = 0; j < 4; ++j) {
    const int   c  = colBase + 16 * j + n;
    const float bj = bias[c];
#pragma unroll
    for (int i = 0; i < 2; ++i) {
#pragma unroll
      for (int v = 0; v < 8; ++v) {
        const int mrow = rowBase + 16 * i + v + half8;
        out[(size_t)mrow * D_ + c] = acc[i][j][v] + bj;
      }
    }
  }
}

// ---------------- launch ----------------
// Workspace layout (f16): hs 16MB | w_in 6MB | w_out 2MB | q 16MB | k 16MB |
// vT 16MB | ctx 16MB  => 88MB total.
extern "C" void kernel_launch(void* const* d_in, const int* in_sizes, int n_in,
                              void* d_out, int out_size, void* d_ws, size_t ws_size,
                              hipStream_t stream) {
  const float* hidden = (const float*)d_in[0];
  const float* w_in   = (const float*)d_in[1];
  const float* b_in   = (const float*)d_in[2];
  const float* w_out  = (const float*)d_in[3];
  const float* b_out  = (const float*)d_in[4];
  float* out = (float*)d_out;

  char* ws = (char*)d_ws;
  _Float16* hs16  = (_Float16*)(ws);
  _Float16* w16   = (_Float16*)(ws + (size_t)(16) * (1 << 20));
  _Float16* wo16  = (_Float16*)(ws + (size_t)(22) * (1 << 20));
  _Float16* q16   = (_Float16*)(ws + (size_t)(24) * (1 << 20));
  _Float16* k16   = (_Float16*)(ws + (size_t)(40) * (1 << 20));
  _Float16* vT16  = (_Float16*)(ws + (size_t)(56) * (1 << 20));
  _Float16* ctx16 = (_Float16*)(ws + (size_t)(72) * (1 << 20));

  const int nhs = B_ * S_ * D_;       // 8388608
  const int nwi = 3 * D_ * D_;        // 3145728
  const int nwo = D_ * D_;            // 1048576
  f32_to_f16_kernel<<<(nhs + 255) / 256, 256, 0, stream>>>(hidden, hs16, nhs);
  f32_to_f16_kernel<<<(nwi + 255) / 256, 256, 0, stream>>>(w_in, w16, nwi);
  f32_to_f16_kernel<<<(nwo + 255) / 256, 256, 0, stream>>>(w_out, wo16, nwo);

  qkv_gemm_kernel<<<dim3(3 * D_ / 128, B_ * S_ / 128), 256, 0, stream>>>(
      hs16, w16, b_in, q16, k16, vT16);

  attn_kernel<<<dim3(B_ * H_, S_ / 64), 128, 0, stream>>>(q16, k16, vT16, ctx16);

  out_gemm_kernel<<<dim3(D_ / 128, B_ * S_ / 128), 256, 0, stream>>>(
      ctx16, wo16, b_out, out);
}
